// MultiHeadAttention_40973988003964
// MI455X (gfx1250) — matrix operand order repaired and checked by reading, not compiled, not run on hardware
//
#include <hip/hip_runtime.h>
#include <hip/hip_bf16.h>

// ---------------------------------------------------------------------------
// MHA forward for MI455X (gfx1250, wave32, WMMA bf16 16x16x32, f32 accum)
// B=2 S=2048 E=1024 H=16 D=64
// Pipeline: f32->bf16 convert pass, LDS-tiled QKV projections,
//           LDS-shared flash attention, LDS-tiled output projection.
// ---------------------------------------------------------------------------

typedef __attribute__((ext_vector_type(16))) __bf16 v16bf;
typedef __attribute__((ext_vector_type(8)))  __bf16 v8bf;
typedef __attribute__((ext_vector_type(4)))  __bf16 v4bf;
typedef __attribute__((ext_vector_type(8)))  float  v8f;

#define S_LEN   2048
#define E_DIM   1024
#define HEADS   16
#define HDIM    64
#define BATCH   2
#define M_ROWS  (BATCH * S_LEN)      // 4096

__device__ __forceinline__ v8f vzero8() {
    v8f z = {0.f, 0.f, 0.f, 0.f, 0.f, 0.f, 0.f, 0.f};
    return z;
}

__device__ __forceinline__ v8f wmma_bf16(v16bf a, v16bf b, v8f c) {
    return __builtin_amdgcn_wmma_f32_16x16x32_bf16(
        /*neg_a=*/false, a, /*neg_b=*/false, b,
        /*c_mod=*/(short)0, c, /*reuse_a=*/false, /*reuse_b=*/false);
}

// A-fragment (16x32, 16-bit): lane half h holds K = h*16..+15
__device__ __forceinline__ v16bf fragA_bf16(const __bf16* row, int k0, int h) {
    v8bf s0 = *(const v8bf*)(row + k0 + h * 16);
    v8bf s1 = *(const v8bf*)(row + k0 + h * 16 + 8);
    v16bf f;
#pragma unroll
    for (int i = 0; i < 8; ++i) { f[i] = s0[i]; f[i + 8] = s1[i]; }
    return f;
}

// ---------------------------------------------------------------------------
// Kernel 0: bulk f32 -> bf16 conversion (bandwidth bound, one pass)
// ---------------------------------------------------------------------------
__global__ __launch_bounds__(256)
void cvt_f32_to_bf16(const float* __restrict__ in, __bf16* __restrict__ out, int n4) {
    const int i = blockIdx.x * 256 + threadIdx.x;
    if (i < n4) {
        const float4 v = ((const float4*)in)[i];
        v4bf o;
        o[0] = (__bf16)v.x; o[1] = (__bf16)v.y;
        o[2] = (__bf16)v.z; o[3] = (__bf16)v.w;
        ((v4bf*)out)[i] = o;
    }
}

// ---------------------------------------------------------------------------
// Kernel 1: Y = bf16(x @ W^T + bias). Block = 128x64 output tile (8 waves x
// one 16x64 tile). Weight 64x32 chunk staged in LDS, shared by all waves.
// layout 0: [B,H,S,D] (Q,K); layout 1: [B,H,D,S] (V transposed).
// ---------------------------------------------------------------------------
__global__ __launch_bounds__(256)
void mha_proj_bf16(const __bf16* __restrict__ X, const __bf16* __restrict__ W,
                   const float* __restrict__ bias, __bf16* __restrict__ Y,
                   int layout) {
    __shared__ __bf16 wbuf[64][32];     // 4 KB weight chunk

    const int wave = threadIdx.x >> 5;
    const int lane = threadIdx.x & 31;
    const int c = lane & 15, h = lane >> 4;

    const int bm = blockIdx.x >> 4;     // 0..31
    const int bn = blockIdx.x & 15;     // 0..15
    const int m0 = bm * 128 + wave * 16;
    const int n0 = bn * 64;

    const __bf16* xrow = X + (size_t)(m0 + c) * E_DIM;

    // cooperative stager mapping: 64 rows x 4 segs of 8 bf16
    const int srow = threadIdx.x >> 2, sseg = threadIdx.x & 3;
    const __bf16* wsrc = W + (size_t)(n0 + srow) * E_DIM + sseg * 8;

    v8f acc[4];
#pragma unroll
    for (int t = 0; t < 4; ++t) acc[t] = vzero8();

    for (int k0 = 0; k0 < E_DIM; k0 += 32) {
        __syncthreads();
        *(v8bf*)&wbuf[srow][sseg * 8] = *(const v8bf*)(wsrc + k0);
        __builtin_prefetch(wsrc + k0 + 32, 0, 1);
        __builtin_prefetch(xrow + k0 + 64, 0, 1);
        __syncthreads();

        v16bf a = fragA_bf16(xrow, k0, h);
#pragma unroll
        for (int t = 0; t < 4; ++t) {
            v16bf b = *(const v16bf*)(&wbuf[t * 16 + c][h * 16]);
            acc[t] = wmma_bf16(a, b, acc[t]);
        }
    }

#pragma unroll
    for (int t = 0; t < 4; ++t) {
        const int n = n0 + t * 16 + c;
        const float bv = bias[n];
        const int hh = n >> 6, d = n & 63;
#pragma unroll
        for (int r = 0; r < 8; ++r) {
            const int m = m0 + r + 8 * h;
            const int b_ = m >> 11, s = m & (S_LEN - 1);
            const float val = acc[t][r] + bv;
            if (layout == 0) {
                Y[(((size_t)b_ * HEADS + hh) * S_LEN + s) * HDIM + d] = (__bf16)val;
            } else {
                Y[(((size_t)b_ * HEADS + hh) * HDIM + d) * S_LEN + s] = (__bf16)val;
            }
        }
    }
}

// ---------------------------------------------------------------------------
// Kernel 2: flash attention. Block = 8 q-blocks of one (b,h); 32-key K and V
// chunks staged in LDS, shared by all 8 waves. Online softmax; P re-laid-out
// C->A through per-wave LDS staging (s_wait_dscnt guarded).
// attn: bf16 [B*S, E] row-major, column = h*64 + d.
// ---------------------------------------------------------------------------
__global__ __launch_bounds__(256)
void mha_flash_attn(const __bf16* __restrict__ Q, const __bf16* __restrict__ K,
                    const __bf16* __restrict__ Vt, __bf16* __restrict__ attn) {
    __shared__ __bf16 kbuf[32][64];     // 4 KB: 32 keys x 64 d
    __shared__ __bf16 vbuf[64][32];     // 4 KB: 64 d x 32 keys
    __shared__ __bf16 pbuf[8][16 * 32]; // 8 KB: per-wave P staging

    const int wave = threadIdx.x >> 5;
    const int lane = threadIdx.x & 31;
    const int c = lane & 15, h = lane >> 4;

    const int bh = blockIdx.x >> 4;                  // 0..31
    const int qb = (blockIdx.x & 15) * 8 + wave;     // 0..127 within (b,h)

    const __bf16* Qbh = Q  + (size_t)bh * S_LEN * HDIM;
    const __bf16* Kbh = K  + (size_t)bh * S_LEN * HDIM;
    const __bf16* Vbh = Vt + (size_t)bh * HDIM * S_LEN;

    // Q A-fragments for d-chunks 0..31, 32..63 (lane row = c)
    const __bf16* qrow = Qbh + (size_t)(qb * 16 + c) * HDIM;
    v16bf qa0 = fragA_bf16(qrow, 0, h);
    v16bf qa1 = fragA_bf16(qrow, 32, h);

    // cooperative stager mappings
    const int kkey = threadIdx.x >> 3, kseg = threadIdx.x & 7;   // K: 32 x 8segs
    const int vd   = threadIdx.x >> 2, vseg = threadIdx.x & 3;   // V: 64 x 4segs

    v8f oacc[4];
#pragma unroll
    for (int t = 0; t < 4; ++t) oacc[t] = vzero8();
    float mrow[8], lrow[8];
#pragma unroll
    for (int r = 0; r < 8; ++r) { mrow[r] = -3.0e38f; lrow[r] = 0.f; }

    const float scale = 0.125f;    // 1/sqrt(64)
    __bf16* pw = &pbuf[wave][0];

    for (int j = 0; j < S_LEN / 32; ++j) {
        const int kbase = j * 32;

        // ---- stage K (32x64) and V (64x32) chunks into LDS ----
        __syncthreads();
        *(v8bf*)&kbuf[kkey][kseg * 8] =
            *(const v8bf*)(Kbh + (size_t)(kbase + kkey) * HDIM + kseg * 8);
        *(v8bf*)&vbuf[vd][vseg * 8] =
            *(const v8bf*)(Vbh + (size_t)vd * S_LEN + kbase + vseg * 8);
        __builtin_prefetch(Kbh + (size_t)(kbase + 32 + kkey) * HDIM + kseg * 8, 0, 1);
        __builtin_prefetch(Vbh + (size_t)vd * S_LEN + kbase + 32 + vseg * 8, 0, 1);
        __syncthreads();

        // ---- scores: 2 C tiles (keys kbase..+16, +16..+32) ----
        v8f s0 = vzero8(), s1 = vzero8();
        {
            v16bf kb0 = *(const v16bf*)(&kbuf[c][h * 16]);
            v16bf kb1 = *(const v16bf*)(&kbuf[16 + c][h * 16]);
            s0 = wmma_bf16(qa0, kb0, s0);
            s1 = wmma_bf16(qa0, kb1, s1);
            kb0 = *(const v16bf*)(&kbuf[c][32 + h * 16]);
            kb1 = *(const v16bf*)(&kbuf[16 + c][32 + h * 16]);
            s0 = wmma_bf16(qa1, kb0, s0);
            s1 = wmma_bf16(qa1, kb1, s1);
        }

        // ---- online softmax over the 32 new columns ----
#pragma unroll
        for (int r = 0; r < 8; ++r) {
            float v0 = s0[r] * scale;
            float v1 = s1[r] * scale;
            float mx = fmaxf(v0, v1);
#pragma unroll
            for (int msk = 1; msk <= 8; msk <<= 1)
                mx = fmaxf(mx, __shfl_xor(mx, msk, 32));
            const float mn = fmaxf(mrow[r], mx);
            const float alpha = __expf(mrow[r] - mn);
            mrow[r] = mn;
            const float p0 = __expf(v0 - mn);
            const float p1 = __expf(v1 - mn);
            s0[r] = p0; s1[r] = p1;
            float rs = p0 + p1;
#pragma unroll
            for (int msk = 1; msk <= 8; msk <<= 1)
                rs += __shfl_xor(rs, msk, 32);
            lrow[r] = lrow[r] * alpha + rs;
#pragma unroll
            for (int t = 0; t < 4; ++t) oacc[t][r] *= alpha;
        }

        // ---- stage P (16x32) row-major in LDS, reload as A-fragment ----
#pragma unroll
        for (int r = 0; r < 8; ++r) {
            pw[(r + 8 * h) * 32 + c]      = (__bf16)s0[r];
            pw[(r + 8 * h) * 32 + 16 + c] = (__bf16)s1[r];
        }
        asm volatile("s_wait_dscnt 0" ::: "memory");
        v16bf pa;
#pragma unroll
        for (int i = 0; i < 8; ++i) {
            pa[i]     = pw[c * 32 + h * 16 + i];
            pa[i + 8] = pw[c * 32 + h * 16 + 8 + i];
        }

        // ---- PV: 4 d-tiles from the shared V chunk ----
#pragma unroll
        for (int t = 0; t < 4; ++t) {
            v16bf vb = *(const v16bf*)(&vbuf[t * 16 + c][h * 16]);
            oacc[t] = wmma_bf16(pa, vb, oacc[t]);
        }
    }

    // ---- finalize: divide by l, write bf16 attn [B*S, E] ----
    float inv[8];
#pragma unroll
    for (int r = 0; r < 8; ++r) inv[r] = 1.0f / lrow[r];
    const int b_ = bh >> 4, hh = bh & 15;
#pragma unroll
    for (int t = 0; t < 4; ++t) {
        const int col = hh * 64 + t * 16 + c;
#pragma unroll
        for (int r = 0; r < 8; ++r) {
            const int s = qb * 16 + r + 8 * h;
            attn[((size_t)b_ * S_LEN + s) * E_DIM + col] = (__bf16)(oacc[t][r] * inv[r]);
        }
    }
}

// ---------------------------------------------------------------------------
// Kernel 3: out(f32) = attn(bf16) @ Wo^T + bo. Same LDS tiling as kernel 1.
// ---------------------------------------------------------------------------
__global__ __launch_bounds__(256)
void mha_out_proj(const __bf16* __restrict__ A, const __bf16* __restrict__ Wo,
                  const float* __restrict__ bo, float* __restrict__ out) {
    __shared__ __bf16 wbuf[64][32];

    const int wave = threadIdx.x >> 5;
    const int lane = threadIdx.x & 31;
    const int c = lane & 15, h = lane >> 4;

    const int bm = blockIdx.x >> 4;
    const int bn = blockIdx.x & 15;
    const int m0 = bm * 128 + wave * 16;
    const int n0 = bn * 64;

    const __bf16* arow = A + (size_t)(m0 + c) * E_DIM;

    const int srow = threadIdx.x >> 2, sseg = threadIdx.x & 3;
    const __bf16* wsrc = Wo + (size_t)(n0 + srow) * E_DIM + sseg * 8;

    v8f acc[4];
#pragma unroll
    for (int t = 0; t < 4; ++t) acc[t] = vzero8();

    for (int k0 = 0; k0 < E_DIM; k0 += 32) {
        __syncthreads();
        *(v8bf*)&wbuf[srow][sseg * 8] = *(const v8bf*)(wsrc + k0);
        __builtin_prefetch(wsrc + k0 + 32, 0, 1);
        __builtin_prefetch(arow + k0 + 64, 0, 1);
        __syncthreads();

        v16bf a = fragA_bf16(arow, k0, h);
#pragma unroll
        for (int t = 0; t < 4; ++t) {
            v16bf b = *(const v16bf*)(&wbuf[t * 16 + c][h * 16]);
            acc[t] = wmma_bf16(a, b, acc[t]);
        }
    }

#pragma unroll
    for (int t = 0; t < 4; ++t) {
        const int n = n0 + t * 16 + c;
        const float bv = bo[n];
#pragma unroll
        for (int r = 0; r < 8; ++r) {
            const int m = m0 + r + 8 * h;
            out[(size_t)m * E_DIM + n] = acc[t][r] + bv;
        }
    }
}

// ---------------------------------------------------------------------------
// Host launcher
// ---------------------------------------------------------------------------
extern "C" void kernel_launch(void* const* d_in, const int* in_sizes, int n_in,
                              void* d_out, int out_size, void* d_ws, size_t ws_size,
                              hipStream_t stream) {
    const float* x  = (const float*)d_in[0];
    const float* Wq = (const float*)d_in[1];
    const float* bq = (const float*)d_in[2];
    const float* Wk = (const float*)d_in[3];
    const float* bk = (const float*)d_in[4];
    const float* Wv = (const float*)d_in[5];
    const float* bv = (const float*)d_in[6];
    const float* Wo = (const float*)d_in[7];
    const float* bo = (const float*)d_in[8];
    float* out = (float*)d_out;

    // Workspace partition (bf16):
    //  xb 8MB | Wqb 2MB | Wkb 2MB | Wvb 2MB | Wob 2MB | qb 8MB | kb 8MB
    //  | vtb 8MB | attn 8MB   (total 48 MB)
    char* ws = (char*)d_ws;
    const size_t MB = 1024ull * 1024ull;
    __bf16* xb   = (__bf16*)(ws);
    __bf16* Wqb  = (__bf16*)(ws + 8 * MB);
    __bf16* Wkb  = (__bf16*)(ws + 10 * MB);
    __bf16* Wvb  = (__bf16*)(ws + 12 * MB);
    __bf16* Wob  = (__bf16*)(ws + 14 * MB);
    __bf16* qb   = (__bf16*)(ws + 16 * MB);
    __bf16* kb   = (__bf16*)(ws + 24 * MB);
    __bf16* vtb  = (__bf16*)(ws + 32 * MB);
    __bf16* attn = (__bf16*)(ws + 40 * MB);

    const dim3 blk(256);

    // f32 -> bf16 conversion passes
    const int nx4 = M_ROWS * E_DIM / 4;       // 1,048,576
    const int nw4 = E_DIM * E_DIM / 4;        //   262,144
    cvt_f32_to_bf16<<<dim3(nx4 / 256), blk, 0, stream>>>(x,  xb,  nx4);
    cvt_f32_to_bf16<<<dim3(nw4 / 256), blk, 0, stream>>>(Wq, Wqb, nw4);
    cvt_f32_to_bf16<<<dim3(nw4 / 256), blk, 0, stream>>>(Wk, Wkb, nw4);
    cvt_f32_to_bf16<<<dim3(nw4 / 256), blk, 0, stream>>>(Wv, Wvb, nw4);
    cvt_f32_to_bf16<<<dim3(nw4 / 256), blk, 0, stream>>>(Wo, Wob, nw4);

    const dim3 grd(512);   // 32 row-blocks x 16 col-blocks (GEMMs) / 32 bh x 16 (attn)
    mha_proj_bf16<<<grd, blk, 0, stream>>>(xb, Wqb, bq, qb, 0);
    mha_proj_bf16<<<grd, blk, 0, stream>>>(xb, Wkb, bk, kb, 0);
    mha_proj_bf16<<<grd, blk, 0, stream>>>(xb, Wvb, bv, vtb, 1);
    mha_flash_attn<<<grd, blk, 0, stream>>>(qb, kb, vtb, attn);
    mha_out_proj<<<grd, blk, 0, stream>>>(attn, Wob, bo, out);
}
